// NODE_11785390260695
// MI455X (gfx1250) — matrix operand order repaired and checked by reading, not compiled, not run on hardware
//
#include <hip/hip_runtime.h>

// ---------------------------------------------------------------------------
// Soft decision forest for MI455X (gfx1250).
//   logits = x(16384x512) @ W^T(384x512)  -> bf16 hi/lo split WMMA, f32 accum
//   then per (row,tree): sigmoid(6), 64 leaf products, dot leaf_weights, sum.
// Round 2: M=64 rows/block, 4 M-tiles per wave -> 4x B-fragment reuse
//          (cuts L2 weight traffic 768MB -> 192MB), K staged in 128-chunks,
//          x-stage LDS aliased with logits LDS, leaf phase in two 32-row passes.
// ---------------------------------------------------------------------------

typedef __attribute__((ext_vector_type(16))) __bf16 v16bf;
typedef __attribute__((ext_vector_type(8)))  __bf16 v8bf;
typedef __attribute__((ext_vector_type(8)))  float  v8f;

#define B_TOTAL 16384
#define D_K     512
#define T_TREES 64
#define DEPTH   6
#define R_COLS  (T_TREES * DEPTH)   // 384 logit columns
#define BM      64                  // rows per block
#define KC      128                 // K chunk staged in LDS
#define XS      (KC + 8)            // 136 bf16 = 68 dwords ≡ 4 mod 64 -> conflict-free
#define LS      388                 // padded f32 logits stride (32 rows per pass)
#define NT      3                   // N-tiles per wave (8 waves x 3 x 16 = 384 cols)
#define MT      4                   // M-tiles per wave (4 x 16 = 64 rows)

// --- split f32 weights into bf16 hi/lo planes (row-major, 384x512) ----------
__global__ void prep_weights(const float* __restrict__ w,
                             __bf16* __restrict__ whi,
                             __bf16* __restrict__ wlo, int n) {
    int i = blockIdx.x * blockDim.x + threadIdx.x;
    if (i < n) {
        float v = w[i];
        __bf16 h = (__bf16)v;            // RNE: top ~8 mantissa bits
        whi[i] = h;
        wlo[i] = (__bf16)(v - (float)h); // next ~8 bits
    }
}

// --- main fused kernel: 64 rows per block, all 384 columns ------------------
__global__ __launch_bounds__(256) void forest_kernel(
    const float*  __restrict__ x,
    const __bf16* __restrict__ whi,
    const __bf16* __restrict__ wlo,
    const float*  __restrict__ thr,   // (T, DEPTH) flat
    const float*  __restrict__ lw,    // (T, 64) flat
    float*        __restrict__ out)   // (B)
{
    // union: x-stage (64 x 136 bf16 hi+lo = 8704 floats) / logits (32 x 388 = 12416 floats)
    __shared__ float smem[32 * LS];
    __shared__ float rowacc[BM];

    __bf16* xhi  = (__bf16*)smem;
    __bf16* xlo  = xhi + BM * XS;
    float*  lgts = smem;

    const int tid     = threadIdx.x;
    const int rowbase = blockIdx.x * BM;

    if (tid < BM) rowacc[tid] = 0.0f;

    const int wave  = tid >> 5;     // 0..7
    const int lane  = tid & 31;
    const int lhalf = lane >> 4;    // 0/1
    const int lmod  = lane & 15;
    const int c0    = wave * (NT * 16);   // column base for this wave

    v8f acc[MT][NT];
#pragma unroll
    for (int mt = 0; mt < MT; ++mt)
#pragma unroll
        for (int j = 0; j < NT; ++j) acc[mt][j] = (v8f){};

    for (int stage = 0; stage < D_K / KC; ++stage) {   // 4 stages of K=128
        __syncthreads();   // previous chunk fully consumed before overwrite
        // ---- stage x chunk (64 rows x 128 k, f32) into LDS as bf16 hi/lo ----
        for (int j = 0; j < (BM * KC) / 256; ++j) {    // 32 iters
            int idx = tid + j * 256;                   // 0..8191
            int r   = idx >> 7;                        // /128
            int k   = idx & (KC - 1);
            float v = x[(size_t)(rowbase + r) * D_K + stage * KC + k];
            __bf16 h = (__bf16)v;
            xhi[r * XS + k] = h;
            xlo[r * XS + k] = (__bf16)(v - (float)h);
        }
        __syncthreads();

        for (int kk = 0; kk < KC / 32; ++kk) {         // 4 K-steps of 32
            // ---- A fragments (16x32 bf16) for 4 M-tiles, hi and lo ----
            const int kb = kk * 32 + lhalf * 16;
            v16bf ahi[MT], alo[MT];
#pragma unroll
            for (int mt = 0; mt < MT; ++mt) {
                const __bf16* ph = &xhi[(mt * 16 + lmod) * XS + kb];
                const __bf16* pl = &xlo[(mt * 16 + lmod) * XS + kb];
                v8bf h0 = *(const v8bf*)ph;
                v8bf h1 = *(const v8bf*)(ph + 8);
                v8bf l0 = *(const v8bf*)pl;
                v8bf l1 = *(const v8bf*)(pl + 8);
#pragma unroll
                for (int e = 0; e < 8; ++e) {
                    ahi[mt][e] = h0[e]; ahi[mt][8 + e] = h1[e];
                    alo[mt][e] = l0[e]; alo[mt][8 + e] = l1[e];
                }
            }
            // ---- B fragments loaded once, reused across 4 M-tiles ----
#pragma unroll
            for (int j = 0; j < NT; ++j) {
                int n = c0 + j * 16 + lmod;            // weight row == logit col
                size_t boff = (size_t)n * D_K + stage * KC + kk * 32 + lhalf * 16;
                v16bf bhi = *(const v16bf*)(whi + boff);
                v16bf blo = *(const v16bf*)(wlo + boff);
#pragma unroll
                for (int mt = 0; mt < MT; ++mt) {
                    acc[mt][j] = __builtin_amdgcn_wmma_f32_16x16x32_bf16(
                        false, ahi[mt], false, bhi, (short)0, acc[mt][j], false, false);
                    acc[mt][j] = __builtin_amdgcn_wmma_f32_16x16x32_bf16(
                        false, ahi[mt], false, blo, (short)0, acc[mt][j], false, false);
                    acc[mt][j] = __builtin_amdgcn_wmma_f32_16x16x32_bf16(
                        false, alo[mt], false, bhi, (short)0, acc[mt][j], false, false);
                }
            }
        }
    }

    // ---- leaf phase: two passes of 32 rows (logits buffer aliases x-stage) ----
#pragma unroll
    for (int pass = 0; pass < 2; ++pass) {
        __syncthreads();   // LDS safe to overwrite (x-stage / previous pass reads done)
        // spill logits for M-tiles 2*pass, 2*pass+1
        // C/D layout: VGPR i, lane L -> M = i + 8*(L>>4), N = L&15
#pragma unroll
        for (int mt2 = 0; mt2 < 2; ++mt2) {
            const int mt = pass * 2 + mt2;
#pragma unroll
            for (int j = 0; j < NT; ++j) {
                int c = c0 + j * 16 + lmod;
#pragma unroll
                for (int i = 0; i < 8; ++i) {
                    int m = mt2 * 16 + i + lhalf * 8;  // 0..31 within pass
                    lgts[m * LS + c] = acc[mt][j][i];
                }
            }
        }
        __syncthreads();

        // 32 rows x 64 trees = 2048 tasks, 8 per thread
        for (int it = 0; it < 8; ++it) {
            int task = tid + it * 256;
            int rl   = task >> 6;       // 0..31; uniform across each wave
            int t    = task & 63;

            float p[DEPTH], q[DEPTH];
#pragma unroll
            for (int d = 0; d < DEPTH; ++d) {
                float z = lgts[rl * LS + t * DEPTH + d] - thr[t * DEPTH + d];
                p[d] = 1.0f / (1.0f + __expf(-z));
                q[d] = 1.0f - p[d];
            }
            // factor 64 leaf products: depths 0-2 (8 combos) x depths 3-5 (8 combos)
            float a[8], b[8];
#pragma unroll
            for (int i = 0; i < 8; ++i) {
                a[i] = ((i & 4) ? q[0] : p[0]) * ((i & 2) ? q[1] : p[1]) *
                       ((i & 1) ? q[2] : p[2]);
                b[i] = ((i & 4) ? q[3] : p[3]) * ((i & 2) ? q[4] : p[4]) *
                       ((i & 1) ? q[5] : p[5]);
            }
            float s = 0.0f;
#pragma unroll
            for (int l = 0; l < 64; ++l)
                s += lw[t * 64 + l] * a[l >> 3] * b[l & 7];

            // wave32 reduction over the 32 trees this wave holds for row rl
            for (int off = 16; off > 0; off >>= 1)
                s += __shfl_xor(s, off, 32);
            if (lane == 0) atomicAdd(&rowacc[pass * 32 + rl], s);  // ds_add_f32
        }
    }
    __syncthreads();

    if (tid < BM) out[rowbase + tid] = rowacc[tid];
}

// ---------------------------------------------------------------------------
extern "C" void kernel_launch(void* const* d_in, const int* in_sizes, int n_in,
                              void* d_out, int out_size, void* d_ws, size_t ws_size,
                              hipStream_t stream) {
    const float* x   = (const float*)d_in[0];   // (16384, 512)
    const float* fs  = (const float*)d_in[1];   // (64, 6, 512)
    const float* thr = (const float*)d_in[2];   // (64, 6)
    const float* lw  = (const float*)d_in[3];   // (64, 64, 1)
    float* out = (float*)d_out;                 // (16384, 1)

    __bf16* whi = (__bf16*)d_ws;                     // 384*512 bf16
    __bf16* wlo = whi + (size_t)R_COLS * D_K;        // 384*512 bf16

    int n = R_COLS * D_K;
    prep_weights<<<(n + 255) / 256, 256, 0, stream>>>(fs, whi, wlo, n);
    forest_kernel<<<B_TOTAL / BM, 256, 0, stream>>>(x, whi, wlo, thr, lw, out);
}
